// TransformerBlock_44822278701283
// MI455X (gfx1250) — matrix operand order repaired and checked by reading, not compiled, not run on hardware
//
#include <hip/hip_runtime.h>
#include <hip/hip_bf16.h>
#include <math.h>

// ---------------------------------------------------------------------------
// CDNA5 (gfx1250) transformer block: bf16 WMMA GEMMs + f32 VALU epilogues.
// wave32; WMMA shape 16x16x32 bf16 -> f32 accumulators.
// Double-buffered LDS staging; async global->LDS (ASYNCcnt) when available.
// ---------------------------------------------------------------------------

typedef __bf16 bf16;
typedef __attribute__((ext_vector_type(16))) __bf16 v16bf;
typedef __attribute__((ext_vector_type(8)))  float  v8f;

#define DIMD    2048
#define NHEADS  16
#define NKVH    4
#define HEADD   128
#define HIDDEN  5632
#define LORAR   8
#define LSCALE  4.0f     // 32/8
#define BB      2
#define SS      2048
#define BSROWS  (BB*SS)  // 4096
#define KVDIM   (NKVH*HEADD) // 512
#define EPSV    1e-5f
#define CHUNKM  1024

static inline int cdiv(long a, long b) { return (int)((a + b - 1) / b); }

// ---------------------------------------------------------------------------
// CDNA5 async global->LDS support (guarded; falls back to sync staging)
// Probe-revealed signature: param0 = int4 AS(1)*, (lds dest, imm offset, cpol)
// ---------------------------------------------------------------------------
#if defined(__gfx1250__) && __has_builtin(__builtin_amdgcn_global_load_async_to_lds_b128)
#define HAVE_ASYNC_LDS 1
typedef __attribute__((ext_vector_type(4))) int i4v;
typedef __attribute__((address_space(1))) i4v gm_i4;
typedef __attribute__((address_space(3))) i4v lds_i4;
#else
#define HAVE_ASYNC_LDS 0
#endif

__device__ __forceinline__ void wait_async_zero() {
#if HAVE_ASYNC_LDS
#if __has_builtin(__builtin_amdgcn_s_wait_asynccnt)
  __builtin_amdgcn_s_wait_asynccnt(0);
#else
  asm volatile("s_wait_asynccnt 0x0" ::: "memory");
#endif
#endif
}

// copy 16 bf16 (32 bytes) global -> LDS
__device__ __forceinline__ void copy16_to_lds(bf16* dst, const bf16* src) {
#if HAVE_ASYNC_LDS
  __builtin_amdgcn_global_load_async_to_lds_b128((gm_i4*)src,       (lds_i4*)dst,       0, 0);
  __builtin_amdgcn_global_load_async_to_lds_b128((gm_i4*)(src + 8), (lds_i4*)(dst + 8), 0, 0);
#else
  *(uint4*)dst       = *(const uint4*)src;
  *(uint4*)(dst + 8) = *(const uint4*)(src + 8);
#endif
}

// ---------------------------------------------------------------------------
// f32 -> bf16 elementwise convert
// ---------------------------------------------------------------------------
__global__ void cvt_f32_bf16_kernel(const float* __restrict__ in,
                                    bf16* __restrict__ out, long n) {
  long i = (long)blockIdx.x * blockDim.x + threadIdx.x;
  if (i < n) out[i] = (bf16)in[i];
}

// ---------------------------------------------------------------------------
// RMSNorm: one block per row of `dim` f32, writes bf16 normalized * w
// ---------------------------------------------------------------------------
__global__ void rmsnorm_bf16_kernel(const float* __restrict__ x,
                                    const float* __restrict__ w,
                                    bf16* __restrict__ out, int dim) {
  const int row = blockIdx.x;
  const float* xr = x + (long)row * dim;
  __shared__ float red[256];
  float ss = 0.f;
  for (int i = threadIdx.x; i < dim; i += 256) { float v = xr[i]; ss += v * v; }
  red[threadIdx.x] = ss;
  __syncthreads();
  for (int s = 128; s > 0; s >>= 1) {
    if (threadIdx.x < (unsigned)s) red[threadIdx.x] += red[threadIdx.x + s];
    __syncthreads();
  }
  const float rinv = rsqrtf(red[0] / (float)dim + EPSV);
  for (int i = threadIdx.x; i < dim; i += 256)
    out[(long)row * dim + i] = (bf16)(xr[i] * rinv * w[i]);
}

// ---------------------------------------------------------------------------
// Tiled WMMA GEMM, double-buffered LDS.
//   OPB==0 : C[M,N] = alpha * A[M,K] * B[N,K]^T   (weights stored row-major [N,K])
//   OPB==1 : C[M,N] = alpha * A[M,K] * B[K,N]     (probs @ V)
// Block tile 128x128, 256 threads = 8 waves (4x2), per-wave 2x4 WMMA tiles.
// BK = 32 == WMMA K. All M,N multiples of 128; K multiple of 32 (no guards).
// Fragment lane mapping per CDNA5 ISA 7.12.2 (bf16 A 16x32 / B 32x16 / C 16x16).
// ---------------------------------------------------------------------------
template <int OPB>
__global__ __launch_bounds__(256) void gemm_bf16_wmma_kernel(
    const bf16* __restrict__ A, const bf16* __restrict__ Bm,
    float* __restrict__ C, int K, int lda, int ldb, int ldc, float alpha) {
  __shared__ __align__(16) bf16 As[2][128][40];
  __shared__ __align__(16) bf16 Bs[2][128][40];

  const int tid  = threadIdx.x;
  const int lane = tid & 31;
  const int wave = tid >> 5;
  const int wm   = wave >> 1;  // 0..3
  const int wn   = wave & 1;   // 0..1
  const int rowBase = blockIdx.y * 128;
  const int colBase = blockIdx.x * 128;

  const int ar = tid >> 1;          // 0..127
  const int ac = (tid & 1) * 16;    // 0 or 16
  const int bk = tid >> 3;          // 0..31   (OPB==1 staging)
  const int bn = (tid & 7) * 16;    // 0..112  (OPB==1 staging)

  v8f acc[2][4];
#pragma unroll
  for (int i = 0; i < 2; ++i)
#pragma unroll
    for (int j = 0; j < 4; ++j) acc[i][j] = (v8f)(0.0f);

  // Stage the 128x32 A tile and 128x32 (as [n][k]) B tile for K-offset kk.
  auto stage = [&](int buf, int kk) {
    copy16_to_lds(&As[buf][ar][ac], A + (long)(rowBase + ar) * lda + kk + ac);
    if (OPB == 0) {
      copy16_to_lds(&Bs[buf][ar][ac], Bm + (long)(colBase + ar) * ldb + kk + ac);
    } else {
      const bf16* src = Bm + (long)(kk + bk) * ldb + colBase + bn;
#pragma unroll
      for (int q = 0; q < 16; ++q) Bs[buf][bn + q][bk] = src[q];
    }
  };

  stage(0, 0);
  wait_async_zero();
  __syncthreads();

  int cur = 0;
  for (int kk = 0; kk < K; kk += 32) {
    const int nxt = cur ^ 1;
    if (kk + 32 < K) {
      stage(nxt, kk + 32);
      if (kk + 64 < K) {  // warm GL2 for the tile after next (global_prefetch)
        __builtin_prefetch(A + (long)(rowBase + ar) * lda + kk + 64 + ac, 0, 1);
        if (OPB == 0)
          __builtin_prefetch(Bm + (long)(colBase + ar) * ldb + kk + 64 + ac, 0, 1);
        else
          __builtin_prefetch(Bm + (long)(kk + 64 + bk) * ldb + colBase + bn, 0, 1);
      }
    }

    // --- build fragments from LDS buffer `cur` ---
    v16bf afrag[2], bfrag[4];
    const int khalfA = (lane >> 4) * 8;
#pragma unroll
    for (int i = 0; i < 2; ++i) {
      const int m = wm * 32 + i * 16 + (lane & 15);
#pragma unroll
      for (int v = 0; v < 8; ++v) {
        const int k0 = 2 * (v & 3) + 16 * (v >> 2) + khalfA;
        afrag[i][2 * v]     = As[cur][m][k0];
        afrag[i][2 * v + 1] = As[cur][m][k0 + 1];
      }
    }
    const int khalfB = (lane >> 4) * 8;
#pragma unroll
    for (int j = 0; j < 4; ++j) {
      const int n = wn * 64 + j * 16 + (lane & 15);
#pragma unroll
      for (int v = 0; v < 8; ++v) {
        bfrag[j][2 * v]     = Bs[cur][n][khalfB + 2 * (v & 3) + 16 * (v >> 2)];
        bfrag[j][2 * v + 1] = Bs[cur][n][khalfB + 2 * (v & 3) + 16 * (v >> 2) + 1];
      }
    }

    // --- 8 WMMAs: D = A*B + C ---
#pragma unroll
    for (int i = 0; i < 2; ++i)
#pragma unroll
      for (int j = 0; j < 4; ++j)
        acc[i][j] = __builtin_amdgcn_wmma_f32_16x16x32_bf16(
            false, afrag[i], false, bfrag[j], (short)0, acc[i][j], false, false);

    wait_async_zero();   // own async fills of buffer `nxt` landed
    __syncthreads();     // all waves done reading `cur` / filling `nxt`
    cur = nxt;
  }

  // --- epilogue: f32 store with alpha ---
#pragma unroll
  for (int i = 0; i < 2; ++i) {
#pragma unroll
    for (int j = 0; j < 4; ++j) {
      const int n = colBase + wn * 64 + j * 16 + (lane & 15);
#pragma unroll
      for (int v = 0; v < 8; ++v) {
        const int m = rowBase + wm * 32 + i * 16 + v + 8 * (lane >> 4);
        C[(long)m * ldc + n] = alpha * acc[i][j][v];
      }
    }
  }
}

// ---------------------------------------------------------------------------
// RoPE (interleaved even/odd pairs), f32 in -> bf16 out. x: [B,S,nh,128]
// ---------------------------------------------------------------------------
__global__ void rope_bf16_kernel(const float* __restrict__ x,
                                 const float* __restrict__ fc,
                                 const float* __restrict__ fs,
                                 bf16* __restrict__ out, int nh) {
  const long npairs = (long)BB * SS * nh * 64;
  long i = (long)blockIdx.x * blockDim.x + threadIdx.x;
  if (i >= npairs) return;
  const int  d  = (int)(i % 64);
  const long t  = i / 64;
  const long bs = t / nh;
  const int  s  = (int)(bs % SS);
  const float c  = fc[(long)s * 64 + d];
  const float sn = fs[(long)s * 64 + d];
  const float xr = x[2 * i], xi = x[2 * i + 1];
  out[2 * i]     = (bf16)(xr * c - xi * sn);
  out[2 * i + 1] = (bf16)(xr * sn + xi * c);
}

// ---------------------------------------------------------------------------
// LoRA down-projection: T[m, 0..7] = X[m,:] @ A[r,:]^T ; one wave per row m.
// ---------------------------------------------------------------------------
__global__ void lora_xa_kernel(const bf16* __restrict__ X,
                               const float* __restrict__ Aw,
                               float* __restrict__ T) {
  const int wave = threadIdx.x >> 5;
  const int lane = threadIdx.x & 31;
  const long m = (long)blockIdx.x * 8 + wave;
  float acc[LORAR];
#pragma unroll
  for (int r = 0; r < LORAR; ++r) acc[r] = 0.f;
  for (int k = lane; k < DIMD; k += 32) {
    const float xv = (float)X[m * DIMD + k];
#pragma unroll
    for (int r = 0; r < LORAR; ++r) acc[r] += xv * Aw[(long)r * DIMD + k];
  }
#pragma unroll
  for (int r = 0; r < LORAR; ++r)
    for (int off = 16; off > 0; off >>= 1)
      acc[r] += __shfl_xor(acc[r], off, 32);
  if (lane < LORAR) T[m * LORAR + lane] = acc[lane];
}

// out[m,n] += tw[m,sel] * LSCALE * (T[m,:] @ Bw[n,:8])
__global__ void lora_add_kernel(float* __restrict__ out,
                                const float* __restrict__ T,
                                const float* __restrict__ Bw,
                                const float* __restrict__ tw,
                                int N, int sel) {
  const long total = (long)BSROWS * N;
  long i = (long)blockIdx.x * blockDim.x + threadIdx.x;
  if (i >= total) return;
  const long m = i / N;
  const int  n = (int)(i % N);
  float s = 0.f;
#pragma unroll
  for (int r = 0; r < LORAR; ++r) s += T[m * LORAR + r] * Bw[(long)n * LORAR + r];
  out[i] += tw[m * 4 + sel] * s * LSCALE;
}

// ---------------------------------------------------------------------------
// Masked softmax over one score row (len S), f32 in -> bf16 probs out.
// ---------------------------------------------------------------------------
__global__ void softmax_mask_kernel(const float* __restrict__ scores,
                                    const float* __restrict__ mask,
                                    bf16* __restrict__ probs) {
  const int s = blockIdx.x;
  const float* row  = scores + (long)s * SS;
  const float* mrow = mask   + (long)s * SS;
  __shared__ float red[256];
  float mx = -1e30f;
  for (int t = threadIdx.x; t < SS; t += 256) mx = fmaxf(mx, row[t] + mrow[t]);
  red[threadIdx.x] = mx; __syncthreads();
  for (int st = 128; st > 0; st >>= 1) {
    if (threadIdx.x < (unsigned)st)
      red[threadIdx.x] = fmaxf(red[threadIdx.x], red[threadIdx.x + st]);
    __syncthreads();
  }
  mx = red[0]; __syncthreads();
  float sum = 0.f;
  for (int t = threadIdx.x; t < SS; t += 256) sum += __expf(row[t] + mrow[t] - mx);
  red[threadIdx.x] = sum; __syncthreads();
  for (int st = 128; st > 0; st >>= 1) {
    if (threadIdx.x < (unsigned)st) red[threadIdx.x] += red[threadIdx.x + st];
    __syncthreads();
  }
  const float inv = 1.f / red[0];
  for (int t = threadIdx.x; t < SS; t += 256)
    probs[(long)s * SS + t] = (bf16)(__expf(row[t] + mrow[t] - mx) * inv);
}

// ---------------------------------------------------------------------------
// Elementwise helpers
// ---------------------------------------------------------------------------
__global__ void add_f32_kernel(const float* __restrict__ a,
                               const float* __restrict__ b,
                               float* __restrict__ o, long n) {
  long i = (long)blockIdx.x * blockDim.x + threadIdx.x;
  if (i < n) o[i] = a[i] + b[i];
}

__global__ void silu_mul_bf16_kernel(const float* __restrict__ g,
                                     const float* __restrict__ u,
                                     bf16* __restrict__ o, long n) {
  long i = (long)blockIdx.x * blockDim.x + threadIdx.x;
  if (i < n) {
    const float gv = g[i];
    o[i] = (bf16)(gv / (1.f + __expf(-gv)) * u[i]);
  }
}

// ---------------------------------------------------------------------------
// Host orchestration
// ---------------------------------------------------------------------------
extern "C" void kernel_launch(void* const* d_in, const int* in_sizes, int n_in,
                              void* d_out, int out_size, void* d_ws, size_t ws_size,
                              hipStream_t stream) {
  (void)in_sizes; (void)n_in; (void)out_size; (void)ws_size;

  const float* x    = (const float*)d_in[0];
  const float* tw   = (const float*)d_in[1];
  const float* fc   = (const float*)d_in[2];
  const float* fs   = (const float*)d_in[3];
  const float* mask = (const float*)d_in[4];
  const float* wq   = (const float*)d_in[6];
  const float* wk   = (const float*)d_in[7];
  const float* wv   = (const float*)d_in[8];
  const float* wo   = (const float*)d_in[9];
  const float* aq   = (const float*)d_in[10];
  const float* bq   = (const float*)d_in[11];
  const float* ak   = (const float*)d_in[12];
  const float* bk   = (const float*)d_in[13];
  const float* av   = (const float*)d_in[14];
  const float* bv   = (const float*)d_in[15];
  const float* ao   = (const float*)d_in[16];
  const float* bo   = (const float*)d_in[17];
  const float* anw  = (const float*)d_in[18];
  const float* fnw  = (const float*)d_in[19];
  const float* w1   = (const float*)d_in[20];
  const float* w2   = (const float*)d_in[21];
  const float* w3   = (const float*)d_in[22];
  float* out = (float*)d_out;

  // Deterministic bump allocator over d_ws.
  size_t off = 0;
  auto alloc = [&](size_t bytes) -> char* {
    char* p = (char*)d_ws + off;
    off = (off + bytes + 255) & ~(size_t)255;
    return p;
  };
  bf16* xnbf   = (bf16*)alloc((size_t)BSROWS * DIMD * 2);
  bf16* wbuf0  = (bf16*)alloc((size_t)HIDDEN * DIMD * 2); // wq/wk/wv/wo then w1
  bf16* wbuf1  = (bf16*)alloc((size_t)HIDDEN * DIMD * 2); // w3
  bf16* wbuf2  = (bf16*)alloc((size_t)DIMD * HIDDEN * 2); // w2
  float* qf    = (float*)alloc((size_t)BSROWS * DIMD * 4);
  float* kf    = (float*)alloc((size_t)BSROWS * KVDIM * 4);
  float* vf    = (float*)alloc((size_t)BSROWS * KVDIM * 4);
  bf16* qbf    = (bf16*)alloc((size_t)BSROWS * DIMD * 2);
  bf16* kbf    = (bf16*)alloc((size_t)BSROWS * KVDIM * 2);
  bf16* vbf    = (bf16*)alloc((size_t)BSROWS * KVDIM * 2);
  float* tq    = (float*)alloc((size_t)BSROWS * LORAR * 4);
  float* tk    = (float*)alloc((size_t)BSROWS * LORAR * 4);
  float* tv    = (float*)alloc((size_t)BSROWS * LORAR * 4);
  float* to    = (float*)alloc((size_t)BSROWS * LORAR * 4);
  float* scoresf = (float*)alloc((size_t)SS * SS * 4);
  bf16* probsbf  = (bf16*)alloc((size_t)SS * SS * 2);
  float* attnf   = (float*)alloc((size_t)BSROWS * DIMD * 4);
  bf16* attnbf   = (bf16*)alloc((size_t)BSROWS * DIMD * 2);
  float* attnout = (float*)alloc((size_t)BSROWS * DIMD * 4);
  float* hbuf    = (float*)alloc((size_t)BSROWS * DIMD * 4);
  bf16* hnbf     = (bf16*)alloc((size_t)BSROWS * DIMD * 2);
  float* gbuf    = (float*)alloc((size_t)CHUNKM * HIDDEN * 4);
  float* ubuf    = (float*)alloc((size_t)CHUNKM * HIDDEN * 4);
  bf16* ffnbf    = (bf16*)alloc((size_t)CHUNKM * HIDDEN * 2);
  float* ffnout  = (float*)alloc((size_t)CHUNKM * DIMD * 4);

  const float SCALEQK = 0.08838834764831845f; // 1/sqrt(128)
  auto cvt = [&](const float* src, bf16* dst, long n) {
    cvt_f32_bf16_kernel<<<cdiv(n, 256), 256, 0, stream>>>(src, dst, n);
  };

  // ---- attention RMSNorm ----
  rmsnorm_bf16_kernel<<<BSROWS, 256, 0, stream>>>(x, anw, xnbf, DIMD);

  // ---- Q ----
  cvt(wq, wbuf0, (long)DIMD * DIMD);
  gemm_bf16_wmma_kernel<0><<<dim3(DIMD / 128, BSROWS / 128), 256, 0, stream>>>(
      xnbf, wbuf0, qf, DIMD, DIMD, DIMD, DIMD, 1.0f);
  lora_xa_kernel<<<BSROWS / 8, 256, 0, stream>>>(xnbf, aq, tq);
  lora_add_kernel<<<cdiv((long)BSROWS * DIMD, 256), 256, 0, stream>>>(qf, tq, bq, tw, DIMD, 0);

  // ---- K ----
  cvt(wk, wbuf0, (long)KVDIM * DIMD);
  gemm_bf16_wmma_kernel<0><<<dim3(KVDIM / 128, BSROWS / 128), 256, 0, stream>>>(
      xnbf, wbuf0, kf, DIMD, DIMD, DIMD, KVDIM, 1.0f);
  lora_xa_kernel<<<BSROWS / 8, 256, 0, stream>>>(xnbf, ak, tk);
  lora_add_kernel<<<cdiv((long)BSROWS * KVDIM, 256), 256, 0, stream>>>(kf, tk, bk, tw, KVDIM, 1);

  // ---- V ----
  cvt(wv, wbuf0, (long)KVDIM * DIMD);
  gemm_bf16_wmma_kernel<0><<<dim3(KVDIM / 128, BSROWS / 128), 256, 0, stream>>>(
      xnbf, wbuf0, vf, DIMD, DIMD, DIMD, KVDIM, 1.0f);
  lora_xa_kernel<<<BSROWS / 8, 256, 0, stream>>>(xnbf, av, tv);
  lora_add_kernel<<<cdiv((long)BSROWS * KVDIM, 256), 256, 0, stream>>>(vf, tv, bv, tw, KVDIM, 2);

  // ---- RoPE + bf16 conversion of V ----
  rope_bf16_kernel<<<cdiv((long)BB * SS * NHEADS * 64, 256), 256, 0, stream>>>(qf, fc, fs, qbf, NHEADS);
  rope_bf16_kernel<<<cdiv((long)BB * SS * NKVH * 64, 256), 256, 0, stream>>>(kf, fc, fs, kbf, NKVH);
  cvt(vf, vbf, (long)BSROWS * KVDIM);

  // ---- attention per (b, h): scores GEMM -> masked softmax -> P@V GEMM ----
  for (int b = 0; b < BB; ++b) {
    for (int h = 0; h < NHEADS; ++h) {
      const bf16* Aq = qbf + (long)b * SS * DIMD + h * HEADD;
      const bf16* Bk = kbf + (long)b * SS * KVDIM + (h / 4) * HEADD;
      gemm_bf16_wmma_kernel<0><<<dim3(SS / 128, SS / 128), 256, 0, stream>>>(
          Aq, Bk, scoresf, HEADD, DIMD, KVDIM, SS, SCALEQK);
      softmax_mask_kernel<<<SS, 256, 0, stream>>>(scoresf, mask, probsbf);
      const bf16* Bv = vbf + (long)b * SS * KVDIM + (h / 4) * HEADD;
      float* Ca = attnf + (long)b * SS * DIMD + h * HEADD;
      gemm_bf16_wmma_kernel<1><<<dim3(HEADD / 128, SS / 128), 256, 0, stream>>>(
          probsbf, Bv, Ca, SS, SS, KVDIM, DIMD, 1.0f);
    }
  }

  // ---- output projection + LoRA-O, residual ----
  cvt(attnf, attnbf, (long)BSROWS * DIMD);
  cvt(wo, wbuf0, (long)DIMD * DIMD);
  gemm_bf16_wmma_kernel<0><<<dim3(DIMD / 128, BSROWS / 128), 256, 0, stream>>>(
      attnbf, wbuf0, attnout, DIMD, DIMD, DIMD, DIMD, 1.0f);
  lora_xa_kernel<<<BSROWS / 8, 256, 0, stream>>>(attnbf, ao, to);
  lora_add_kernel<<<cdiv((long)BSROWS * DIMD, 256), 256, 0, stream>>>(attnout, to, bo, tw, DIMD, 3);
  add_f32_kernel<<<cdiv((long)BSROWS * DIMD, 256), 256, 0, stream>>>(x, attnout, hbuf, (long)BSROWS * DIMD);

  // ---- FFN ----
  rmsnorm_bf16_kernel<<<BSROWS, 256, 0, stream>>>(hbuf, fnw, hnbf, DIMD);
  cvt(w1, wbuf0, (long)HIDDEN * DIMD);
  cvt(w3, wbuf1, (long)HIDDEN * DIMD);
  cvt(w2, wbuf2, (long)DIMD * HIDDEN);
  for (int c = 0; c < BSROWS / CHUNKM; ++c) {
    const bf16* Ah = hnbf + (long)c * CHUNKM * DIMD;
    gemm_bf16_wmma_kernel<0><<<dim3(HIDDEN / 128, CHUNKM / 128), 256, 0, stream>>>(
        Ah, wbuf0, gbuf, DIMD, DIMD, DIMD, HIDDEN, 1.0f);
    gemm_bf16_wmma_kernel<0><<<dim3(HIDDEN / 128, CHUNKM / 128), 256, 0, stream>>>(
        Ah, wbuf1, ubuf, DIMD, DIMD, DIMD, HIDDEN, 1.0f);
    silu_mul_bf16_kernel<<<cdiv((long)CHUNKM * HIDDEN, 256), 256, 0, stream>>>(
        gbuf, ubuf, ffnbf, (long)CHUNKM * HIDDEN);
    gemm_bf16_wmma_kernel<0><<<dim3(DIMD / 128, CHUNKM / 128), 256, 0, stream>>>(
        ffnbf, wbuf2, ffnout, HIDDEN, HIDDEN, HIDDEN, DIMD, 1.0f);
    add_f32_kernel<<<cdiv((long)CHUNKM * DIMD, 256), 256, 0, stream>>>(
        hbuf + (long)c * CHUNKM * DIMD, ffnout, out + (long)c * CHUNKM * DIMD,
        (long)CHUNKM * DIMD);
  }
}
